// LargePredCodingQA_70360154243364
// MI455X (gfx1250) — matrix operand order repaired and checked by reading, not compiled, not run on hardware
//
#include <hip/hip_runtime.h>
#include <hip/hip_bf16.h>
#include <math.h>

// Problem constants (match reference)
#define Bm 128
#define Sm 384
#define Dm 768
#define Hm 3072
#define Lm 12
#define NITER 20

typedef __attribute__((ext_vector_type(16))) __bf16 v16bf;
typedef __attribute__((ext_vector_type(8)))  float  v8f;
typedef __attribute__((ext_vector_type(4)))  unsigned int v4u;

union Frag16 { v16bf h; v4u q[2]; };

// vector-of-4-int type matching the async-LDS builtin's expected pointee
typedef int v4i_t __attribute__((vector_size(16)));
typedef __attribute__((address_space(1))) v4i_t* as1_v4i;
typedef __attribute__((address_space(3))) v4i_t* as3_v4i;

__device__ __forceinline__ unsigned short f2bf(float x) {
  unsigned int u = __float_as_uint(x);
  u += 0x7FFFu + ((u >> 16) & 1u);   // round-to-nearest-even
  return (unsigned short)(u >> 16);
}
__device__ __forceinline__ float gelu_exact(float x) {
  return 0.5f * x * (1.0f + erff(x * 0.70710678118654752440f));
}

// ---- CDNA5 async global->LDS copy (ASYNCcnt-tracked), 16B per lane ----
__device__ __forceinline__ void async_copy_b128(const unsigned short* gsrc,
                                                unsigned short* lds_generic) {
#if __has_builtin(__builtin_amdgcn_global_load_async_to_lds_b128)
  as1_v4i g = (as1_v4i)(unsigned long long)(size_t)gsrc;                 // AS1 value == generic
  as3_v4i l = (as3_v4i)(unsigned int)(size_t)lds_generic;                // LDS addr = low 32 bits
  __builtin_amdgcn_global_load_async_to_lds_b128(g, l, 0, 0);
#else
  unsigned int loff = (unsigned int)(size_t)lds_generic;
  asm volatile("global_load_async_to_lds_b128 %0, %1, off"
               :: "v"(loff), "v"(gsrc) : "memory");
#endif
}
__device__ __forceinline__ void async_wait0() {
#if __has_builtin(__builtin_amdgcn_s_wait_asynccnt)
  __builtin_amdgcn_s_wait_asynccnt(0);
#else
  asm volatile("s_wait_asynccnt 0x0" ::: "memory");
#endif
}

// ---------------- fp32 -> bf16 bulk convert ----------------
__global__ void k_f32_to_bf16(const float* __restrict__ src,
                              unsigned short* __restrict__ dst, long n) {
  long i = (long)blockIdx.x * blockDim.x + threadIdx.x;
  long stride = (long)gridDim.x * blockDim.x;
  for (; i < n; i += stride) dst[i] = f2bf(src[i]);
}

// ---------------- embeddings + LayerNorm ----------------
__global__ void k_embed_ln(const int* __restrict__ ids, const float* __restrict__ we,
                           const float* __restrict__ pe, const float* __restrict__ te,
                           const float* __restrict__ g, const float* __restrict__ b,
                           float* __restrict__ out) {
  int tok = blockIdx.x;               // b*S + s
  int s = tok % Sm;
  int id = ids[tok];
  const float* wrow = we + (long)id * Dm;
  const float* prow = pe + (long)s * Dm;
  __shared__ float red[256];
  float v[3]; float sum = 0.f;
  for (int e = 0; e < 3; ++e) {
    int d = threadIdx.x + e * 256;
    v[e] = wrow[d] + prow[d] + te[d];
    sum += v[e];
  }
  red[threadIdx.x] = sum; __syncthreads();
  for (int o = 128; o; o >>= 1) { if (threadIdx.x < o) red[threadIdx.x] += red[threadIdx.x + o]; __syncthreads(); }
  float mean = red[0] * (1.0f / Dm); __syncthreads();
  float vs = 0.f;
  for (int e = 0; e < 3; ++e) { float t = v[e] - mean; vs += t * t; }
  red[threadIdx.x] = vs; __syncthreads();
  for (int o = 128; o; o >>= 1) { if (threadIdx.x < o) red[threadIdx.x] += red[threadIdx.x + o]; __syncthreads(); }
  float inv = rsqrtf(red[0] * (1.0f / Dm) + 1e-5f);
  float* orow = out + (long)tok * Dm;
  for (int e = 0; e < 3; ++e) {
    int d = threadIdx.x + e * 256;
    orow[d] = (v[e] - mean) * inv * g[d] + b[d];
  }
}

// ---------------- masked mean pooling over S ----------------
__global__ void k_pool(const float* __restrict__ embeds, const int* __restrict__ mask,
                       float* __restrict__ pooled) {
  int b = blockIdx.x; int d = threadIdx.x;   // 768 threads
  float acc = 0.f, ms = 0.f;
  const float* base = embeds + (long)b * Sm * Dm + d;
  const int* mrow = mask + b * Sm;
  for (int s = 0; s < Sm; ++s) {
    float mf = (float)mrow[s];
    acc += mf * base[(long)s * Dm];
    ms += mf;
  }
  pooled[b * Dm + d] = acc / fmaxf(ms, 1.0f);
}

// ---------------- WMMA bf16 GEMM, M=128 fixed, block tile 128x64 ----------------
// C = A[128,K] @ B[K,N] + bias; epilogue: gelu_bf16 ? bf16(GELU(.)) : fp32(.)
// Batched over blockIdx.z. B tiles stream via async global->LDS, double-buffered.
__global__ void __launch_bounds__(256)
k_gemm_bf16(const unsigned short* __restrict__ A, const unsigned short* __restrict__ Bw,
            const float* __restrict__ bias, void* __restrict__ Cout,
            int N, int K, long sA, long sB, long sBias, long sC, int gelu_bf16) {
  const int l = blockIdx.z;
  A    += (long)l * sA;
  Bw   += (long)l * sB;
  bias += (long)l * sBias;
  const int n0   = blockIdx.x * 64;
  const int tid  = threadIdx.x;
  const int lane = tid & 31;
  const int wave = tid >> 5;
  const int r0   = wave * 16;                 // 8 waves x 16-row strips = 128 rows
  __shared__ __align__(16) unsigned short Bs[2][32][72]; // double buffer, +8 pad

  v8f acc[4];
#pragma unroll
  for (int j = 0; j < 4; ++j)
#pragma unroll
    for (int e = 0; e < 8; ++e) acc[j][e] = 0.f;

  const int kr = tid >> 3;            // 0..31  (staging row)
  const int nc = (tid & 7) * 8;       // 0..56  (staging col, 16B chunks)
  const int arow  = r0 + (lane & 15); // A row per ISA 16-bit A layout
  const int kbase = (lane >> 4) * 8;  // lanes 0-15: K {0..7,16..23}; 16-31: {8..15,24..31}

  // prologue: stage tile 0 asynchronously
  async_copy_b128(Bw + (long)kr * N + n0 + nc, &Bs[0][kr][nc]);

  int buf = 0;
  for (int k0 = 0; k0 < K; k0 += 32) {
    async_wait0();        // own async writes to Bs[buf] landed in LDS
    __syncthreads();      // all waves' tile data visible; prev reads of buf^1 done

    if (k0 + 32 < K)      // overlap next tile's fetch with this tile's compute
      async_copy_b128(Bw + (long)(k0 + 32 + kr) * N + n0 + nc, &Bs[buf ^ 1][kr][nc]);

    Frag16 fa;
    const unsigned short* ap = A + (long)arow * K + k0 + kbase;
    fa.q[0] = *(const v4u*)ap;          // K = kbase..kbase+7
    fa.q[1] = *(const v4u*)(ap + 16);   // K = kbase+16..kbase+23

#pragma unroll
    for (int j = 0; j < 4; ++j) {
      Frag16 fb;                         // B fragment: N = j*16 + lane%16, K = kbase..+7, kbase+16..+23
      const unsigned short* bp = &Bs[buf][kbase][j * 16 + (lane & 15)];
#pragma unroll
      for (int s = 0; s < 8; ++s) {
        fb.h[s]     = __builtin_bit_cast(__bf16, bp[s * 72]);
        fb.h[s + 8] = __builtin_bit_cast(__bf16, bp[(s + 16) * 72]);
      }
      acc[j] = __builtin_amdgcn_wmma_f32_16x16x32_bf16(
                   false, fa.h, false, fb.h, (short)0, acc[j], false, false);
    }
    buf ^= 1;
  }

  // Epilogue: C layout — VGPR i: lanes 0-15 -> row i, lanes 16-31 -> row i+8; col = lane&15
  const int colbase = n0 + (lane & 15);
  const int rowadd  = (lane >> 4) * 8;
  if (gelu_bf16) {
    unsigned short* Cb = (unsigned short*)Cout + (long)l * sC;
#pragma unroll
    for (int j = 0; j < 4; ++j) {
      int col = colbase + j * 16;
      float bv = bias[col];
#pragma unroll
      for (int i = 0; i < 8; ++i) {
        int row = r0 + rowadd + i;
        Cb[(long)row * N + col] = f2bf(gelu_exact(acc[j][i] + bv));
      }
    }
  } else {
    float* Cf = (float*)Cout + (long)l * sC;
#pragma unroll
    for (int j = 0; j < 4; ++j) {
      int col = colbase + j * 16;
      float bv = bias[col];
#pragma unroll
      for (int i = 0; i < 8; ++i) {
        int row = r0 + rowadd + i;
        Cf[(long)row * N + col] = acc[j][i] + bv;
      }
    }
  }
}

// ---------------- fused residual (+scale) + LayerNorm, fp32 + bf16 out ----------------
__global__ void k_residual_ln(const float* __restrict__ res, const float* __restrict__ y,
                              const float* __restrict__ step, const float* __restrict__ g,
                              const float* __restrict__ bb, float* __restrict__ outf,
                              unsigned short* __restrict__ outh,
                              long sR, long sY, long sG, long sO) {
  int l = blockIdx.z, row = blockIdx.x;
  const float* rr = res + (long)l * sR + (long)row * Dm;
  const float* yy = y   + (long)l * sY + (long)row * Dm;
  const float* gg = g   + (long)l * sG;
  const float* be = bb  + (long)l * sG;
  float s = step ? step[l] : 1.0f;
  __shared__ float red[256];
  float t[3]; float sum = 0.f;
  for (int e = 0; e < 3; ++e) {
    int d = threadIdx.x + e * 256;
    t[e] = rr[d] + s * yy[d];
    sum += t[e];
  }
  red[threadIdx.x] = sum; __syncthreads();
  for (int o = 128; o; o >>= 1) { if (threadIdx.x < o) red[threadIdx.x] += red[threadIdx.x + o]; __syncthreads(); }
  float mean = red[0] * (1.0f / Dm); __syncthreads();
  float vs = 0.f;
  for (int e = 0; e < 3; ++e) { float q = t[e] - mean; vs += q * q; }
  red[threadIdx.x] = vs; __syncthreads();
  for (int o = 128; o; o >>= 1) { if (threadIdx.x < o) red[threadIdx.x] += red[threadIdx.x + o]; __syncthreads(); }
  float inv = rsqrtf(red[0] * (1.0f / Dm) + 1e-5f);
  float* of = outf + (long)l * sO + (long)row * Dm;
  unsigned short* oh = outh + (long)l * sO + (long)row * Dm;
  for (int e = 0; e < 3; ++e) {
    int d = threadIdx.x + e * 256;
    float v = (t[e] - mean) * inv * gg[d] + be[d];
    of[d] = v; oh[d] = f2bf(v);
  }
}

// ---------------- prediction-error combine (writes bf16 for next GEMM) ----------------
__global__ void k_err(const float* __restrict__ reps, const float* __restrict__ preds,
                      const float* __restrict__ pooled, unsigned short* __restrict__ totalh) {
  const long BD = (long)Bm * Dm;
  long n = (long)Lm * BD;
  long i = (long)blockIdx.x * blockDim.x + threadIdx.x;
  long stride = (long)gridDim.x * blockDim.x;
  for (; i < n; i += stride) {
    long l = i / BD, r = i % BD;
    float below = (l == 0 ? pooled[r] : reps[(l - 1) * BD + r]) - preds[i];
    float above = (l < Lm - 1) ? (reps[i] - preds[(l + 1) * BD + r]) : 0.f;
    totalh[i] = f2bf(below + 0.5f * above);
  }
}

__global__ void k_add(const float* __restrict__ a, const float* __restrict__ b,
                      float* __restrict__ c, int n) {
  int i = blockIdx.x * blockDim.x + threadIdx.x;
  if (i < n) c[i] = a[i] + b[i];
}

// ---------------- small fp32 GEMM (heads, <1% of FLOPs) ----------------
__global__ void k_gemm_f32(const float* __restrict__ A, const float* __restrict__ Bmat,
                           const float* __restrict__ bias, float* __restrict__ C,
                           int M, int N, int K, int transB) {
  int i = blockIdx.x * blockDim.x + threadIdx.x;
  if (i >= M * N) return;
  int m = i / N, n = i % N;
  const float* ar = A + (long)m * K;
  float acc = bias ? bias[n] : 0.f;
  if (transB) {
    const float* br = Bmat + (long)n * K;
    for (int k = 0; k < K; ++k) acc += ar[k] * br[k];
  } else {
    for (int k = 0; k < K; ++k) acc += ar[k] * Bmat[(long)k * N + n];
  }
  C[i] = acc;
}

// c[b] = q[b,:] . v  (tk_b bias folded into pointer score)
__global__ void k_dotvec(const float* __restrict__ q, const float* __restrict__ v,
                         float* __restrict__ out) {
  int b = blockIdx.x;
  __shared__ float red[256];
  float s = 0.f;
  for (int e = 0; e < 3; ++e) { int d = threadIdx.x + e * 256; s += q[(long)b * Dm + d] * v[d]; }
  red[threadIdx.x] = s; __syncthreads();
  for (int o = 128; o; o >>= 1) { if (threadIdx.x < o) red[threadIdx.x] += red[threadIdx.x + o]; __syncthreads(); }
  if (!threadIdx.x) out[b] = red[0];
}

// Pointer logits via algebraic fold: logits[b,s] = (embeds[b,s,:].u[b] + c[b]) / (NH*sqrt(HD))
__global__ void k_logits(const float* __restrict__ embeds, const float* __restrict__ us,
                         const float* __restrict__ ue, const float* __restrict__ cs,
                         const float* __restrict__ ce, const int* __restrict__ mask,
                         float* __restrict__ sl, float* __restrict__ el) {
  int tok = blockIdx.x;             // b*S + s
  int b = tok / Sm;
  const float* er = embeds + (long)tok * Dm;
  const float* u1 = us + (long)b * Dm;
  const float* u2 = ue + (long)b * Dm;
  __shared__ float r1[256], r2[256];
  float a1 = 0.f, a2 = 0.f;
  for (int e = 0; e < 3; ++e) {
    int d = threadIdx.x + e * 256;
    float x = er[d];
    a1 += x * u1[d]; a2 += x * u2[d];
  }
  r1[threadIdx.x] = a1; r2[threadIdx.x] = a2; __syncthreads();
  for (int o = 128; o; o >>= 1) {
    if (threadIdx.x < o) { r1[threadIdx.x] += r1[threadIdx.x + o]; r2[threadIdx.x] += r2[threadIdx.x + o]; }
    __syncthreads();
  }
  if (!threadIdx.x) {
    const float scale = 1.0f / (12.0f * 8.0f);  // 1/(NH*sqrt(HD))
    bool m = mask[tok] != 0;
    sl[tok] = m ? (r1[0] + cs[b]) * scale : -1e9f;
    el[tok] = m ? (r2[0] + ce[b]) * scale : -1e9f;
  }
}

// ans head: GELU(refined @ w1 + b1) @ w2 + b2  -> [B,2]
__global__ void k_ans(const float* __restrict__ refined, const float* __restrict__ w1,
                      const float* __restrict__ b1, const float* __restrict__ w2,
                      const float* __restrict__ b2, float* __restrict__ out) {
  int b = blockIdx.x;
  __shared__ float h[384];
  const float* x = refined + (long)b * Dm;
  int j = threadIdx.x;               // 384 threads
  float acc = b1[j];
  for (int d = 0; d < Dm; ++d) acc += x[d] * w1[(long)d * 384 + j];
  h[j] = gelu_exact(acc);
  __syncthreads();
  if (j < 2) {
    float o = b2[j];
    for (int k = 0; k < 384; ++k) o += h[k] * w2[k * 2 + j];
    out[b * 2 + j] = o;
  }
}

extern "C" void kernel_launch(void* const* d_in, const int* in_sizes, int n_in,
                              void* d_out, int out_size, void* d_ws, size_t ws_size,
                              hipStream_t stream) {
  (void)in_sizes; (void)n_in; (void)out_size; (void)ws_size;
  const int*   input_ids = (const int*)d_in[0];
  const int*   attn_mask = (const int*)d_in[1];
  const float* word_emb  = (const float*)d_in[2];
  const float* pos_emb   = (const float*)d_in[3];
  const float* type_emb  = (const float*)d_in[4];
  const float* emb_g     = (const float*)d_in[5];
  const float* emb_b     = (const float*)d_in[6];
  const float* ff_w1     = (const float*)d_in[7];
  const float* ff_b1     = (const float*)d_in[8];
  const float* ff_w2     = (const float*)d_in[9];
  const float* ff_b2     = (const float*)d_in[10];
  const float* ffn_g     = (const float*)d_in[11];
  const float* ffn_b     = (const float*)d_in[12];
  const float* pd_w1     = (const float*)d_in[13];
  const float* pd_b1     = (const float*)d_in[14];
  const float* pd_w2     = (const float*)d_in[15];
  const float* pd_b2     = (const float*)d_in[16];
  const float* pe_w1     = (const float*)d_in[17];
  const float* pe_b1     = (const float*)d_in[18];
  const float* pe_w2     = (const float*)d_in[19];
  const float* pe_b2     = (const float*)d_in[20];
  const float* un_g      = (const float*)d_in[21];
  const float* un_b      = (const float*)d_in[22];
  const float* step      = (const float*)d_in[23];
  const float* sq_w      = (const float*)d_in[24];
  const float* sq_b      = (const float*)d_in[25];
  const float* eq_w      = (const float*)d_in[26];
  const float* eq_b      = (const float*)d_in[27];
  const float* tk_w      = (const float*)d_in[28];
  const float* tk_b      = (const float*)d_in[29];
  const float* ans_w1    = (const float*)d_in[30];
  const float* ans_b1    = (const float*)d_in[31];
  const float* ans_w2    = (const float*)d_in[32];
  const float* ans_b2    = (const float*)d_in[33];

  // ---- workspace carve (256B aligned) ----
  char* wp = (char*)d_ws;
  size_t off = 0;
  auto carve = [&](size_t bytes) -> char* {
    char* p = wp + off;
    off = (off + bytes + 255) & ~(size_t)255;
    return p;
  };
  const long BD  = (long)Bm * Dm;            // 98304
  const long LBD = (long)Lm * BD;
  const long WEL = (long)Lm * Dm * Hm;       // 28,311,552 per weight tensor

  float*          embeds   = (float*)carve((long)Bm * Sm * Dm * 4);   // 151 MB
  float*          pooled   = (float*)carve(BD * 4);
  unsigned short* pooled_h = (unsigned short*)carve(BD * 2);
  float*          reps     = (float*)carve(LBD * 4);
  unsigned short* reps_h   = (unsigned short*)carve(LBD * 2);
  unsigned short* hidden_h = (unsigned short*)carve((long)Lm * Bm * Hm * 2);
  float*          t1       = (float*)carve(LBD * 4);                  // preds / pe-out
  unsigned short* total_h  = (unsigned short*)carve(LBD * 2);
  float*          ff_out   = (float*)carve(BD * 4);
  float*          refined  = (float*)carve(BD * 4);
  float*          start_q  = (float*)carve(BD * 4);
  float*          end_q    = (float*)carve(BD * 4);
  float*          u_s      = (float*)carve(BD * 4);
  float*          u_e      = (float*)carve(BD * 4);
  float*          c_s      = (float*)carve(Bm * 4);
  float*          c_e      = (float*)carve(Bm * 4);
  unsigned short* ffw1b    = (unsigned short*)carve(WEL * 2);
  unsigned short* ffw2b    = (unsigned short*)carve(WEL * 2);
  unsigned short* pdw1b    = (unsigned short*)carve(WEL * 2);
  unsigned short* pdw2b    = (unsigned short*)carve(WEL * 2);
  unsigned short* pew1b    = (unsigned short*)carve(WEL * 2);
  unsigned short* pew2b    = (unsigned short*)carve(WEL * 2);

  // 1) one-time (per call) bf16 weight conversion: halves hot-loop HBM traffic
  k_f32_to_bf16<<<4096, 256, 0, stream>>>(ff_w1, ffw1b, WEL);
  k_f32_to_bf16<<<4096, 256, 0, stream>>>(ff_w2, ffw2b, WEL);
  k_f32_to_bf16<<<4096, 256, 0, stream>>>(pd_w1, pdw1b, WEL);
  k_f32_to_bf16<<<4096, 256, 0, stream>>>(pd_w2, pdw2b, WEL);
  k_f32_to_bf16<<<4096, 256, 0, stream>>>(pe_w1, pew1b, WEL);
  k_f32_to_bf16<<<4096, 256, 0, stream>>>(pe_w2, pew2b, WEL);

  // 2) embeddings + LN, masked mean pooling
  k_embed_ln<<<Bm * Sm, 256, 0, stream>>>(input_ids, word_emb, pos_emb, type_emb, emb_g, emb_b, embeds);
  k_pool<<<Bm, Dm, 0, stream>>>(embeds, attn_mask, pooled);
  k_f32_to_bf16<<<96, 256, 0, stream>>>(pooled, pooled_h, BD);

  // 3) init scan: cur = LN(cur + MLP_ff(cur)) sequentially through L layers
  for (int l = 0; l < Lm; ++l) {
    const unsigned short* Ain = l ? reps_h + (long)(l - 1) * BD : pooled_h;
    k_gemm_bf16<<<dim3(Hm / 64, 1, 1), 256, 0, stream>>>(
        Ain, ffw1b + (long)l * Dm * Hm, ff_b1 + (long)l * Hm,
        hidden_h, Hm, Dm, 0, 0, 0, 0, 1);
    k_gemm_bf16<<<dim3(Dm / 64, 1, 1), 256, 0, stream>>>(
        hidden_h, ffw2b + (long)l * Hm * Dm, ff_b2 + (long)l * Dm,
        t1, Dm, Hm, 0, 0, 0, 0, 0);
    const float* rres = l ? reps + (long)(l - 1) * BD : pooled;
    k_residual_ln<<<dim3(Bm, 1, 1), 256, 0, stream>>>(
        rres, t1, nullptr, ffn_g + (long)l * Dm, ffn_b + (long)l * Dm,
        reps + (long)l * BD, reps_h + (long)l * BD, 0, 0, 0, 0);
  }
  (void)hipMemcpyAsync(ff_out, reps + (long)(Lm - 1) * BD, BD * 4, hipMemcpyDeviceToDevice, stream);

  // 4) 20 predictive-coding iterations (all GEMMs batched over L via grid.z)
  for (int it = 0; it < NITER; ++it) {
    // preds = MLP_pd(reps)
    k_gemm_bf16<<<dim3(Hm / 64, 1, Lm), 256, 0, stream>>>(
        reps_h, pdw1b, pd_b1, hidden_h, Hm, Dm, BD, (long)Dm * Hm, Hm, (long)Bm * Hm, 1);
    k_gemm_bf16<<<dim3(Dm / 64, 1, Lm), 256, 0, stream>>>(
        hidden_h, pdw2b, pd_b2, t1, Dm, Hm, (long)Bm * Hm, (long)Hm * Dm, Dm, BD, 0);
    // total = err_below + 0.5*err_above
    k_err<<<1152, 256, 0, stream>>>(reps, t1, pooled, total_h);
    // reps = LN(reps + step * MLP_pe(total))
    k_gemm_bf16<<<dim3(Hm / 64, 1, Lm), 256, 0, stream>>>(
        total_h, pew1b, pe_b1, hidden_h, Hm, Dm, BD, (long)Dm * Hm, Hm, (long)Bm * Hm, 1);
    k_gemm_bf16<<<dim3(Dm / 64, 1, Lm), 256, 0, stream>>>(
        hidden_h, pew2b, pe_b2, t1, Dm, Hm, (long)Bm * Hm, (long)Hm * Dm, Dm, BD, 0);
    k_residual_ln<<<dim3(Bm, 1, Lm), 256, 0, stream>>>(
        reps, t1, step, un_g, un_b, reps, reps_h, BD, BD, Dm, BD);
  }

  // 5) heads
  k_add<<<(int)(BD / 256), 256, 0, stream>>>(reps + (long)(Lm - 1) * BD, ff_out, refined, (int)BD);
  k_gemm_f32<<<(Bm * Dm + 255) / 256, 256, 0, stream>>>(refined, sq_w, sq_b, start_q, Bm, Dm, Dm, 0);
  k_gemm_f32<<<(Bm * Dm + 255) / 256, 256, 0, stream>>>(refined, eq_w, eq_b, end_q, Bm, Dm, Dm, 0);
  // fold tk_w into query: u = q @ tk_w^T ; c = q . tk_b
  k_gemm_f32<<<(Bm * Dm + 255) / 256, 256, 0, stream>>>(start_q, tk_w, nullptr, u_s, Bm, Dm, Dm, 1);
  k_gemm_f32<<<(Bm * Dm + 255) / 256, 256, 0, stream>>>(end_q, tk_w, nullptr, u_e, Bm, Dm, Dm, 1);
  k_dotvec<<<Bm, 256, 0, stream>>>(start_q, tk_b, c_s);
  k_dotvec<<<Bm, 256, 0, stream>>>(end_q, tk_b, c_e);

  float* outp = (float*)d_out;
  k_logits<<<Bm * Sm, 256, 0, stream>>>(embeds, u_s, u_e, c_s, c_e, attn_mask,
                                        outp, outp + (long)Bm * Sm);
  k_ans<<<Bm, 384, 0, stream>>>(refined, ans_w1, ans_b1, ans_w2, ans_b2, outp + 2l * Bm * Sm);
}
